// Attention_26371099198429
// MI455X (gfx1250) — matrix operand order repaired and checked by reading, not compiled, not run on hardware
//
#include <hip/hip_runtime.h>
#include <hip/hip_bf16.h>

// ---------- types ----------
typedef __attribute__((ext_vector_type(16))) __bf16 v16bf;
typedef __attribute__((ext_vector_type(8)))  float  v8f;

struct __align__(16) U4 { unsigned int x, y, z, w; };
struct __align__(16) U8 { U4 lo, hi; };

// v16bf operand: first 8 elems from p0, last 8 from p1 (two b128 loads).
__device__ __forceinline__ v16bf load_pair(const unsigned short* p0,
                                           const unsigned short* p1) {
  U8 u;
  u.lo = *reinterpret_cast<const U4*>(p0);
  u.hi = *reinterpret_cast<const U4*>(p1);
  return __builtin_bit_cast(v16bf, u);
}
// 16 bf16 as p[0..7], p[16..23] (table k order) — B-operand pattern.
__device__ __forceinline__ v16bf load16(const unsigned short* p) {
  return load_pair(p, p + 16);
}
// All-ones bf16 operand (for row-sum WMMA).
__device__ __forceinline__ v16bf ones16() {
  U8 u;
  u.lo.x = u.lo.y = u.lo.z = u.lo.w = 0x3F803F80u;
  u.hi = u.lo;
  return __builtin_bit_cast(v16bf, u);
}

__device__ __forceinline__ unsigned short f2bf(float f) {
  unsigned int x = __builtin_bit_cast(unsigned int, f);
  x += 0x7FFFu + ((x >> 16) & 1u);          // round-to-nearest-even
  return (unsigned short)(x >> 16);
}

__device__ __forceinline__ v8f wmma_bf16(v16bf a, v16bf b, v8f c) {
  return __builtin_amdgcn_wmma_f32_16x16x32_bf16(false, a, false, b,
                                                 (short)0, c, false, false);
}

#if __has_builtin(__builtin_amdgcn_exp2f)
#define EXP2F(x) __builtin_amdgcn_exp2f(x)
#else
#define EXP2F(x) exp2f(x)
#endif

#define LOG2E 1.44269504088896340736f
#define NTOK  4096
#define CDIM  256

// ---------- kernel 0: weights fp32 -> bf16 ----------
__global__ void cvt_w_kernel(const float* __restrict__ wqkv,
                             const float* __restrict__ wproj,
                             unsigned short* __restrict__ wqkv_bf,
                             unsigned short* __restrict__ wproj_bf) {
  int i = blockIdx.x * 256 + threadIdx.x;
  if (i < 768 * 256) wqkv_bf[i] = f2bf(wqkv[i]);
  if (i < 256 * 256) wproj_bf[i] = f2bf(wproj[i]);
}

// ---------- kernel 1: LayerNorm over C, emit bf16 tokens [B*N, C] ----------
__global__ void ln_kernel(const float* __restrict__ x,
                          const float* __restrict__ gamma,
                          const float* __restrict__ beta,
                          unsigned short* __restrict__ tn) {
  int idx = blockIdx.x * blockDim.x + threadIdx.x;   // 0..B*N-1
  int b = idx >> 12, n = idx & (NTOK - 1);
  const float* xp = x + (size_t)b * CDIM * NTOK + n; // x[b, c, n], stride NTOK
  float s = 0.f, s2 = 0.f;
  for (int c = 0; c < CDIM; ++c) {
    float v = xp[(size_t)c * NTOK];
    s += v; s2 += v * v;
  }
  float mu   = s * (1.0f / CDIM);
  float var  = s2 * (1.0f / CDIM) - mu * mu;
  float rstd = rsqrtf(var + 1e-5f);
  unsigned short* tp = tn + (size_t)idx * CDIM;
  for (int c = 0; c < CDIM; ++c) {
    float v = (xp[(size_t)c * NTOK] - mu) * rstd * gamma[c] + beta[c];
    tp[c] = f2bf(v);
  }
}

// ---------- kernel 2: QKV GEMM (8192x256 @ 256x768), 16x128 tile per wave ----
// q,k: [b,h,n,d] bf16 (q pre-scaled by 1/sqrt(d)); v transposed vt: [b,h,d,n].
__global__ void qkv_kernel(const unsigned short* __restrict__ tn,
                           const unsigned short* __restrict__ wqkv,
                           unsigned short* __restrict__ q,
                           unsigned short* __restrict__ k,
                           unsigned short* __restrict__ vt) {
  int wave = threadIdx.x >> 5, lane = threadIdx.x & 31;
  int lrow = lane & 15, lhalf = lane >> 4;
  int m0 = blockIdx.x * 128 + wave * 16;
  int n0 = blockIdx.y * 128;

  const unsigned short* arow = tn + (size_t)(m0 + lrow) * CDIM + lhalf * 8;
  const unsigned short* brow[8];
#pragma unroll
  for (int t = 0; t < 8; ++t)
    brow[t] = wqkv + (size_t)(n0 + t * 16 + lrow) * CDIM + lhalf * 8;

  v8f acc[8] = {};
#pragma unroll
  for (int k0 = 0; k0 < CDIM; k0 += 32) {
    v16bf a = load_pair(arow + k0, arow + k0 + 16);
    v16bf bt[8];
#pragma unroll
    for (int t = 0; t < 8; ++t) bt[t] = load16(brow[t] + k0);
#pragma unroll
    for (int t = 0; t < 8; ++t) acc[t] = wmma_bf16(a, bt[t], acc[t]);
  }

  const float scale = 0.17677669529663687f;  // 1/sqrt(32)
#pragma unroll
  for (int t = 0; t < 8; ++t) {
    int o = n0 + t * 16 + lrow;
    int kind = o >> 8;          // 0=q, 1=k, 2=v
    int oc = o & 255;
    int h = oc >> 5, d = oc & 31;
#pragma unroll
    for (int r = 0; r < 8; ++r) {
      int row = m0 + lhalf * 8 + r;            // global token row
      int b = row >> 12, n = row & (NTOK - 1);
      size_t bh = (size_t)(b * 8 + h);
      float v = acc[t][r];
      if (kind == 0)      q[(bh * NTOK + n) * 32 + d] = f2bf(v * scale);
      else if (kind == 1) k[(bh * NTOK + n) * 32 + d] = f2bf(v);
      else                vt[(bh * 32 + d) * NTOK + n] = f2bf(v);
    }
  }
}

// ---------- kernel 3: fused flash attention ----------
// Two 16-row Q tiles per wave share each 64-wide K/V chunk (halves K/V load
// traffic, two independent WMMA chains). Row sums via WMMA against all-ones.
__global__ void attn_kernel(const unsigned short* __restrict__ q,
                            const unsigned short* __restrict__ k,
                            const unsigned short* __restrict__ vt,
                            unsigned short* __restrict__ ao) {
  __shared__ unsigned short plds[8][2][16 * 64];   // 32 KB: P tiles per wave
  int wave = threadIdx.x >> 5, lane = threadIdx.x & 31;
  int lrow = lane & 15, lhalf = lane >> 4;
  int bh = blockIdx.y;                              // 0..15
  int b = bh >> 3, h = bh & 7;
  int nb0 = blockIdx.x * 256 + wave * 16;           // Q tile 0
  int nb1 = nb0 + 128;                              // Q tile 1

  const unsigned short* qb = q  + (size_t)bh * NTOK * 32;
  const unsigned short* kb = k  + (size_t)bh * NTOK * 32;
  const unsigned short* vb = vt + (size_t)bh * 32 * NTOK;

  // Resident Q A-operands (16 rows x 32 d each), already scaled.
  const unsigned short* qp0 = qb + (size_t)(nb0 + lrow) * 32 + lhalf * 8;
  const unsigned short* qp1 = qb + (size_t)(nb1 + lrow) * 32 + lhalf * 8;
  v16bf aq0 = load_pair(qp0, qp0 + 16);
  v16bf aq1 = load_pair(qp1, qp1 + 16);
  v16bf bones = ones16();

  v8f o00 = {}, o01 = {}, ow0 = {};
  v8f o10 = {}, o11 = {}, ow1 = {};
  float mr0[8], mr1[8];
#pragma unroll
  for (int r = 0; r < 8; ++r) { mr0[r] = -1e30f; mr1[r] = -1e30f; }
  v8f vzero = {};

  for (int m0 = 0; m0 < NTOK; m0 += 64) {
    // ---- K and V operands for this chunk (shared by both Q tiles) ----
    v16bf bk[4], bv[4];
#pragma unroll
    for (int i = 0; i < 4; ++i)
      bk[i] = load16(kb + (size_t)(m0 + i * 16 + lrow) * 32 + lhalf * 8);
#pragma unroll
    for (int j = 0; j < 2; ++j)
#pragma unroll
      for (int i = 0; i < 2; ++i)
        bv[j * 2 + i] =
            load16(vb + (size_t)(i * 16 + lrow) * NTOK + m0 + j * 32 + lhalf * 8);
    if (m0 + 64 < NTOK) {                         // deep prefetch of next chunk
      __builtin_prefetch(kb + (size_t)(m0 + 64 + lrow * 4) * 32, 0, 0);
      __builtin_prefetch(vb + (size_t)lrow * NTOK + m0 + 64, 0, 0);
    }

    // ---- S = Q K^T for both Q tiles ----
    v8f sA[4], sB[4];
#pragma unroll
    for (int i = 0; i < 4; ++i) sA[i] = wmma_bf16(aq0, bk[i], vzero);
#pragma unroll
    for (int i = 0; i < 4; ++i) sB[i] = wmma_bf16(aq1, bk[i], vzero);

    // ---- online softmax for tile 0 ----
#pragma unroll
    for (int r = 0; r < 8; ++r) {
      float cm = fmaxf(fmaxf(sA[0][r], sA[1][r]), fmaxf(sA[2][r], sA[3][r]));
      cm = fmaxf(cm, __shfl_xor(cm, 1, 32));
      cm = fmaxf(cm, __shfl_xor(cm, 2, 32));
      cm = fmaxf(cm, __shfl_xor(cm, 4, 32));
      cm = fmaxf(cm, __shfl_xor(cm, 8, 32));
      float nm    = fmaxf(mr0[r], cm);
      float alpha = EXP2F((mr0[r] - nm) * LOG2E);
      float p0 = EXP2F((sA[0][r] - nm) * LOG2E);
      float p1 = EXP2F((sA[1][r] - nm) * LOG2E);
      float p2 = EXP2F((sA[2][r] - nm) * LOG2E);
      float p3 = EXP2F((sA[3][r] - nm) * LOG2E);
      o00[r] *= alpha; o01[r] *= alpha; ow0[r] *= alpha;
      mr0[r] = nm;
      int row = lhalf * 8 + r;
      plds[wave][0][row * 64 +  0 + lrow] = f2bf(p0);
      plds[wave][0][row * 64 + 16 + lrow] = f2bf(p1);
      plds[wave][0][row * 64 + 32 + lrow] = f2bf(p2);
      plds[wave][0][row * 64 + 48 + lrow] = f2bf(p3);
    }
    // ---- online softmax for tile 1 ----
#pragma unroll
    for (int r = 0; r < 8; ++r) {
      float cm = fmaxf(fmaxf(sB[0][r], sB[1][r]), fmaxf(sB[2][r], sB[3][r]));
      cm = fmaxf(cm, __shfl_xor(cm, 1, 32));
      cm = fmaxf(cm, __shfl_xor(cm, 2, 32));
      cm = fmaxf(cm, __shfl_xor(cm, 4, 32));
      cm = fmaxf(cm, __shfl_xor(cm, 8, 32));
      float nm    = fmaxf(mr1[r], cm);
      float alpha = EXP2F((mr1[r] - nm) * LOG2E);
      float p0 = EXP2F((sB[0][r] - nm) * LOG2E);
      float p1 = EXP2F((sB[1][r] - nm) * LOG2E);
      float p2 = EXP2F((sB[2][r] - nm) * LOG2E);
      float p3 = EXP2F((sB[3][r] - nm) * LOG2E);
      o10[r] *= alpha; o11[r] *= alpha; ow1[r] *= alpha;
      mr1[r] = nm;
      int row = lhalf * 8 + r;
      plds[wave][1][row * 64 +  0 + lrow] = f2bf(p0);
      plds[wave][1][row * 64 + 16 + lrow] = f2bf(p1);
      plds[wave][1][row * 64 + 32 + lrow] = f2bf(p2);
      plds[wave][1][row * 64 + 48 + lrow] = f2bf(p3);
    }
    asm volatile("s_wait_dscnt 0" ::: "memory");  // P stores -> A-layout reloads

    const unsigned short* pp0 = &plds[wave][0][lrow * 64 + lhalf * 8];
    const unsigned short* pp1 = &plds[wave][1][lrow * 64 + lhalf * 8];
    v16bf a00 = load_pair(pp0, pp0 + 16);         // tile0, P cols  0..31
    v16bf a01 = load_pair(pp0 + 32, pp0 + 48);    // tile0, P cols 32..63
    v16bf a10 = load_pair(pp1, pp1 + 16);         // tile1, P cols  0..31
    v16bf a11 = load_pair(pp1 + 32, pp1 + 48);    // tile1, P cols 32..63

    // ---- O += P @ V ; row-sums += P @ ones (both tiles) ----
    o00 = wmma_bf16(a00, bv[0], o00);
    o10 = wmma_bf16(a10, bv[0], o10);
    o01 = wmma_bf16(a00, bv[1], o01);
    o11 = wmma_bf16(a10, bv[1], o11);
    ow0 = wmma_bf16(a00, bones, ow0);
    ow1 = wmma_bf16(a10, bones, ow1);
    o00 = wmma_bf16(a01, bv[2], o00);
    o10 = wmma_bf16(a11, bv[2], o10);
    o01 = wmma_bf16(a01, bv[3], o01);
    o11 = wmma_bf16(a11, bv[3], o11);
    ow0 = wmma_bf16(a01, bones, ow0);
    ow1 = wmma_bf16(a11, bones, ow1);
  }

  // Normalize (ow holds row sums replicated in every column) and store bf16.
#pragma unroll
  for (int r = 0; r < 8; ++r) {
    float inv0 = 1.0f / ow0[r];
    float inv1 = 1.0f / ow1[r];
    int n_0 = nb0 + lhalf * 8 + r;
    int n_1 = nb1 + lhalf * 8 + r;
    size_t base0 = ((size_t)b * NTOK + n_0) * CDIM + h * 32;
    size_t base1 = ((size_t)b * NTOK + n_1) * CDIM + h * 32;
    ao[base0 + lrow]      = f2bf(o00[r] * inv0);
    ao[base0 + 16 + lrow] = f2bf(o01[r] * inv0);
    ao[base1 + lrow]      = f2bf(o10[r] * inv1);
    ao[base1 + 16 + lrow] = f2bf(o11[r] * inv1);
  }
}

// ---------- kernel 4: proj GEMM + bias + residual, 16x128 tile per wave -----
__global__ void proj_kernel(const unsigned short* __restrict__ ao,
                            const unsigned short* __restrict__ wproj,
                            const float* __restrict__ bproj,
                            const float* __restrict__ x,
                            float* __restrict__ out) {
  int wave = threadIdx.x >> 5, lane = threadIdx.x & 31;
  int lrow = lane & 15, lhalf = lane >> 4;
  int m0 = blockIdx.x * 128 + wave * 16;
  int n0 = blockIdx.y * 128;

  const unsigned short* arow = ao + (size_t)(m0 + lrow) * CDIM + lhalf * 8;
  const unsigned short* brow[8];
#pragma unroll
  for (int t = 0; t < 8; ++t)
    brow[t] = wproj + (size_t)(n0 + t * 16 + lrow) * CDIM + lhalf * 8;

  v8f acc[8] = {};
#pragma unroll
  for (int k0 = 0; k0 < CDIM; k0 += 32) {
    v16bf a = load_pair(arow + k0, arow + k0 + 16);
    v16bf bt[8];
#pragma unroll
    for (int t = 0; t < 8; ++t) bt[t] = load16(brow[t] + k0);
#pragma unroll
    for (int t = 0; t < 8; ++t) acc[t] = wmma_bf16(a, bt[t], acc[t]);
  }

#pragma unroll
  for (int t = 0; t < 8; ++t) {
    int o = n0 + t * 16 + lrow;
    float bias = bproj[o];
    int rowbase = m0 + lhalf * 8;
    int b = rowbase >> 12, n = rowbase & (NTOK - 1);
    const float* xp = x   + ((size_t)b * CDIM + o) * NTOK + n;
    float*       op = out + ((size_t)b * CDIM + o) * NTOK + n;
#pragma unroll
    for (int r = 0; r < 8; ++r) op[r] = acc[t][r] + bias + xp[r];  // contiguous
  }
}

// ---------- launch ----------
extern "C" void kernel_launch(void* const* d_in, const int* in_sizes, int n_in,
                              void* d_out, int out_size, void* d_ws, size_t ws_size,
                              hipStream_t stream) {
  const float* x     = (const float*)d_in[0];
  const float* gamma = (const float*)d_in[1];
  const float* beta  = (const float*)d_in[2];
  const float* wqkv  = (const float*)d_in[3];
  const float* wproj = (const float*)d_in[4];
  const float* bproj = (const float*)d_in[5];
  float* out = (float*)d_out;

  char* ws = (char*)d_ws;
  const size_t TOK_BYTES = (size_t)2 * NTOK * CDIM * 2;   // 4 MB bf16 [B*N, C]
  unsigned short* tn       = (unsigned short*)(ws);
  unsigned short* q        = (unsigned short*)(ws + TOK_BYTES);
  unsigned short* k        = (unsigned short*)(ws + 2 * TOK_BYTES);
  unsigned short* vt       = (unsigned short*)(ws + 3 * TOK_BYTES);
  unsigned short* ao       = (unsigned short*)(ws + 4 * TOK_BYTES);
  unsigned short* wqkv_bf  = (unsigned short*)(ws + 5 * TOK_BYTES);
  unsigned short* wproj_bf = (unsigned short*)(ws + 5 * TOK_BYTES + 768 * 256 * 2);

  cvt_w_kernel<<<768, 256, 0, stream>>>(wqkv, wproj, wqkv_bf, wproj_bf);
  ln_kernel<<<(2 * NTOK) / 256, 256, 0, stream>>>(x, gamma, beta, tn);
  qkv_kernel<<<dim3(8192 / 128, 768 / 128), 256, 0, stream>>>(tn, wqkv_bf, q, k, vt);
  attn_kernel<<<dim3(NTOK / 256, 16), 256, 0, stream>>>(q, k, vt, ao);
  proj_kernel<<<dim3(8192 / 128, 256 / 128), 256, 0, stream>>>(ao, wproj_bf, bproj, x, out);
}
